// MultiHeadAttentionBase_65481071409777
// MI455X (gfx1250) — matrix operand order repaired and checked by reading, not compiled, not run on hardware
//
#include <hip/hip_runtime.h>

typedef unsigned short u16;
typedef __bf16 bf16_t;
typedef __attribute__((ext_vector_type(16))) bf16_t v16bf;
typedef __attribute__((ext_vector_type(16))) u16    v16u;
typedef __attribute__((ext_vector_type(8)))  float  v8f;

#define EMBED 1024
#define SEQ   2048
#define BATCH 4
#define HEADS 16
#define HDIM  64

__device__ __forceinline__ u16 f2bf(float f) {
  unsigned u = __builtin_bit_cast(unsigned, f);
  u += 0x7FFFu + ((u >> 16) & 1u);   // round-to-nearest-even
  return (u16)(u >> 16);
}

__device__ __forceinline__ v8f wmma_bf16(v16u a, v16u b, v8f c) {
  return __builtin_amdgcn_wmma_f32_16x16x32_bf16(
      false, __builtin_bit_cast(v16bf, a),
      false, __builtin_bit_cast(v16bf, b),
      (short)0, c, false, false);
}

// async global->LDS b128 pair: copies 32B (two b128) from gaddr to LDS offset laddr.
// ISA: INST_OFFSET is added to BOTH the LDS and the memory address.
__device__ __forceinline__ void async_copy32(unsigned laddr, unsigned long long gaddr) {
  asm volatile(
      "global_load_async_to_lds_b128 %0, %1, off\n\t"
      "global_load_async_to_lds_b128 %0, %1, off offset:16"
      :: "v"(laddr), "v"(gaddr) : "memory");
}
__device__ __forceinline__ void async_copy16(unsigned laddr, unsigned long long gaddr) {
  asm volatile("global_load_async_to_lds_b128 %0, %1, off"
               :: "v"(laddr), "v"(gaddr) : "memory");
}
__device__ __forceinline__ unsigned lds_off(const void* p) {
  return (unsigned)(uintptr_t)p;     // LDS aperture: addr[31:0] is the LDS offset
}

// ---------------- fp32 -> bf16 convert ----------------
__global__ void cvt_f32_bf16(const float* __restrict__ src, u16* __restrict__ dst, int n4) {
  int i = blockIdx.x * blockDim.x + threadIdx.x;
  int stride = gridDim.x * blockDim.x;
  for (; i < n4; i += stride) {
    float4 v = reinterpret_cast<const float4*>(src)[i];
    ushort4 o;
    o.x = f2bf(v.x); o.y = f2bf(v.y); o.z = f2bf(v.z); o.w = f2bf(v.w);
    reinterpret_cast<ushort4*>(dst)[i] = o;
  }
}

// ---------------- Y = X @ W^T (+bias), bf16 WMMA, async double-buffered ----------------
// X: [M,K] bf16 row-major; W: [N,K] bf16 row-major (torch Linear weight).
// Block: 64x64 output tile, 128 threads = 4 waves, each wave 32x32 (2x2 wmma).
template <bool F32OUT>
__global__ __launch_bounds__(128)
void gemm_bf16w(const u16* __restrict__ X, const u16* __restrict__ W,
                u16* __restrict__ Ybf, float* __restrict__ Yf,
                const float* __restrict__ bias, int Mdim, int Ndim, int Kdim) {
  __shared__ __align__(16) u16 Xs[2][64 * 32];
  __shared__ __align__(16) u16 Ws[2][64 * 32];

  const int tid  = threadIdx.x;
  const int lane = tid & 31;
  const int wave = tid >> 5;
  const int l15  = lane & 15;
  const int hi   = lane >> 4;

  const int nBase = blockIdx.x * 64;
  const int mBase = blockIdx.y * 64;
  const int mW = (wave >> 1) * 32;
  const int nW = (wave & 1) * 32;

  const int rowL = tid >> 1;           // 0..63
  const int colL = (tid & 1) * 16;     // 0 or 16 (ushorts)

  v8f acc[2][2];
#pragma unroll
  for (int i = 0; i < 2; i++)
#pragma unroll
    for (int j = 0; j < 2; j++)
#pragma unroll
      for (int r = 0; r < 8; r++) acc[i][j][r] = 0.0f;

  const unsigned lXrow = lds_off(&Xs[0][0]) + (unsigned)(rowL * 64 + colL * 2);
  const unsigned lWrow = lds_off(&Ws[0][0]) + (unsigned)(rowL * 64 + colL * 2);
  const size_t gXrow = (size_t)(mBase + rowL) * Kdim + colL;
  const size_t gWrow = (size_t)(nBase + rowL) * Kdim + colL;

  // prologue: stage kk=0 into buffer 0  (4 async b128 per thread per stage)
  async_copy32(lXrow, (unsigned long long)(X + gXrow));
  async_copy32(lWrow, (unsigned long long)(W + gWrow));

  int buf = 0;
  for (int kk = 0; kk < Kdim; kk += 32, buf ^= 1) {
    if (kk + 32 < Kdim) {
      unsigned nb = (unsigned)((buf ^ 1) * 64 * 32 * 2);
      async_copy32(lXrow + nb, (unsigned long long)(X + gXrow + kk + 32));
      async_copy32(lWrow + nb, (unsigned long long)(W + gWrow + kk + 32));
      asm volatile("s_wait_asynccnt 0x4" ::: "memory");   // current buffer done
    } else {
      asm volatile("s_wait_asynccnt 0x0" ::: "memory");
    }
    __syncthreads();

    const u16* Xb = &Xs[buf][0];
    const u16* Wb = &Ws[buf][0];
    v16u a[2], b[2];
#pragma unroll
    for (int i = 0; i < 2; i++) {
      int row = mW + i * 16 + l15;
      int kb = hi * 8;                       // A-frag: two 8-wide K runs
#pragma unroll
      for (int h = 0; h < 16; h++) {
        int k = kb + (h < 8 ? h : 8 + h);
        a[i][h] = Xb[row * 32 + k];
      }
    }
#pragma unroll
    for (int j = 0; j < 2; j++) {
      int nrow = nW + j * 16 + l15;
      int kb2 = hi * 8;                      // B-frag: n=lane&15, two 8-wide K runs
#pragma unroll
      for (int h = 0; h < 16; h++) b[j][h] = Wb[nrow * 32 + kb2 + (h < 8 ? h : 8 + h)];
    }
#pragma unroll
    for (int i = 0; i < 2; i++)
#pragma unroll
      for (int j = 0; j < 2; j++)
        acc[i][j] = wmma_bf16(a[i], b[j], acc[i][j]);
    __syncthreads();                         // safe to overwrite this buffer next round
  }

#pragma unroll
  for (int i = 0; i < 2; i++)
#pragma unroll
    for (int j = 0; j < 2; j++)
#pragma unroll
      for (int r = 0; r < 8; r++) {
        int row = mBase + mW + i * 16 + r + hi * 8;
        int col = nBase + nW + j * 16 + l15;
        float v = acc[i][j][r];
        if (F32OUT) Yf[(size_t)row * Ndim + col] = v + bias[col];
        else        Ybf[(size_t)row * Ndim + col] = f2bf(v);
      }
}

// ---------------- flash attention, one wave per 16-query tile ----------------
__global__ __launch_bounds__(256)
void mha_attn(const u16* __restrict__ Qb, const u16* __restrict__ Kb,
              const u16* __restrict__ Vb, u16* __restrict__ Cb) {
  __shared__ __align__(16) u16 Ks[2][32 * 64];
  __shared__ __align__(16) u16 Vs[2][32 * 64];
  __shared__ __align__(16) u16 Ps[8 * 16 * 32];   // per-wave P staging

  const int tid  = threadIdx.x;
  const int lane = tid & 31;
  const int wave = tid >> 5;
  const int l15  = lane & 15;
  const int hi   = lane >> 4;

  const int b = blockIdx.z, h = blockIdx.y;
  const int qBase = blockIdx.x * 128 + wave * 16;

  const size_t bhOff = (size_t)b * SEQ * EMBED + (size_t)h * HDIM;
  const u16* Q = Qb + bhOff;
  const u16* K = Kb + bhOff;
  const u16* V = Vb + bhOff;
  u16* C = Cb + bhOff;

  // scores in log2 domain: raw * (1/sqrt(64) * log2(e)) so exp == v_exp_f32
  const float SCL2 = 0.125f * 1.44269504088896f;

  // Q A-fragments over d-chunks {0..31},{32..63}; loaded once
  v16u qa[2];
#pragma unroll
  for (int c = 0; c < 2; c++) {
    int row = qBase + l15;
    int kb = hi * 8;
#pragma unroll
    for (int hh = 0; hh < 16; hh++) {
      int d = c * 32 + kb + (hh < 8 ? hh : 8 + hh);
      qa[c][hh] = Q[(size_t)row * EMBED + d];
    }
  }

  // all-ones bf16 B fragment for row-sum-via-wmma
  v16u onesB;
#pragma unroll
  for (int hh = 0; hh < 16; hh++) onesB[hh] = 0x3F80;

  float Mr[8], Lr[8], corr[8];
  v8f o[4];
#pragma unroll
  for (int r = 0; r < 8; r++) { Mr[r] = -1e30f; Lr[r] = 0.0f; }
#pragma unroll
  for (int t = 0; t < 4; t++)
#pragma unroll
    for (int r = 0; r < 8; r++) o[t][r] = 0.0f;

  const int rowT = tid >> 3;
  const int colT = (tid & 7) * 8;
  u16* Pw = Ps + wave * (16 * 32);

  const unsigned lK = lds_off(&Ks[0][0]) + (unsigned)(tid * 16);
  const unsigned lV = lds_off(&Vs[0][0]) + (unsigned)(tid * 16);
  const size_t gRow = (size_t)rowT * EMBED + colT;

  // prologue: stage kv=0 into buffer 0  (2 async b128 per thread per stage)
  async_copy16(lK, (unsigned long long)(K + gRow));
  async_copy16(lV, (unsigned long long)(V + gRow));

  int buf = 0;
  for (int kv = 0; kv < SEQ; kv += 32, buf ^= 1) {
    if (kv + 32 < SEQ) {
      unsigned nb = (unsigned)((buf ^ 1) * 32 * 64 * 2);
      size_t g = gRow + (size_t)(kv + 32) * EMBED;
      async_copy16(lK + nb, (unsigned long long)(K + g));
      async_copy16(lV + nb, (unsigned long long)(V + g));
      asm volatile("s_wait_asynccnt 0x2" ::: "memory");
    } else {
      asm volatile("s_wait_asynccnt 0x0" ::: "memory");
    }
    __syncthreads();

    const u16* Kt = &Ks[buf][0];
    const u16* Vt = &Vs[buf][0];

    // scores: S[sub] = Q(16x64) . K_sub^T(64x16), two chained wmmas each
    v8f s[2];
#pragma unroll
    for (int sub = 0; sub < 2; sub++) {
#pragma unroll
      for (int r = 0; r < 8; r++) s[sub][r] = 0.0f;
#pragma unroll
      for (int c = 0; c < 2; c++) {
        v16u bk;
        int key = sub * 16 + l15;
#pragma unroll
        for (int hh = 0; hh < 16; hh++)
          bk[hh] = Kt[key * 64 + c * 32 + hi * 8 + (hh < 8 ? hh : 8 + hh)];
        s[sub] = wmma_bf16(qa[c], bk, s[sub]);
      }
    }

    // online softmax (log2 domain); row max via 16-lane xor-shuffles
#pragma unroll
    for (int r = 0; r < 8; r++) {
      float s0 = s[0][r] * SCL2;
      float s1 = s[1][r] * SCL2;
      float rm = fmaxf(s0, s1);
#pragma unroll
      for (int m = 1; m < 16; m <<= 1) rm = fmaxf(rm, __shfl_xor(rm, m, 32));
      float nM = fmaxf(Mr[r], rm);
      corr[r] = exp2f(Mr[r] - nM);
      float p0 = exp2f(s0 - nM);
      float p1 = exp2f(s1 - nM);
      Mr[r] = nM;
#pragma unroll
      for (int t = 0; t < 4; t++) o[t][r] *= corr[r];
      int rowp = r + hi * 8;                // C-layout -> LDS (row-major P)
      Pw[rowp * 32 + l15]      = f2bf(p0);
      Pw[rowp * 32 + 16 + l15] = f2bf(p1);
    }

    asm volatile("s_wait_dscnt 0x0" ::: "memory");  // LDS RAW within wave

    // re-read P in A-fragment layout
    v16u pa;
    {
      int kb = hi * 8;
#pragma unroll
      for (int hh = 0; hh < 16; hh++) {
        int k = kb + (hh < 8 ? hh : 8 + hh);
        pa[hh] = Pw[l15 * 32 + k];
      }
    }

    // row sums of P via one wmma against all-ones B (replaces 32 bpermutes)
    v8f rs;
#pragma unroll
    for (int r = 0; r < 8; r++) rs[r] = 0.0f;
    rs = wmma_bf16(pa, onesB, rs);
#pragma unroll
    for (int r = 0; r < 8; r++) Lr[r] = Lr[r] * corr[r] + rs[r];

    // O += P(16x32) . V(32x64): 4 wmmas over d-tiles
#pragma unroll
    for (int t = 0; t < 4; t++) {
      v16u bv;
#pragma unroll
      for (int hh = 0; hh < 16; hh++)
        bv[hh] = Vt[(hi * 8 + (hh < 8 ? hh : 8 + hh)) * 64 + t * 16 + l15];
      o[t] = wmma_bf16(pa, bv, o[t]);
    }
    __syncthreads();                        // safe to overwrite this buffer next round
  }

  // finalize: divide by L, write ctx as bf16 in [B*S, E] with head offset
#pragma unroll
  for (int r = 0; r < 8; r++) {
    float inv = 1.0f / Lr[r];
    int row = qBase + r + hi * 8;
#pragma unroll
    for (int t = 0; t < 4; t++)
      C[(size_t)row * EMBED + t * 16 + l15] = f2bf(o[t][r] * inv);
  }
}

extern "C" void kernel_launch(void* const* d_in, const int* in_sizes, int n_in,
                              void* d_out, int out_size, void* d_ws, size_t ws_size,
                              hipStream_t stream) {
  (void)in_sizes; (void)n_in; (void)out_size; (void)ws_size;
  const float* x  = (const float*)d_in[0];
  const float* Wq = (const float*)d_in[1];
  const float* Wk = (const float*)d_in[2];
  const float* Wv = (const float*)d_in[3];
  const float* Wo = (const float*)d_in[4];
  const float* bo = (const float*)d_in[5];

  const int M = BATCH * SEQ;   // 8192
  const int E = EMBED;         // 1024

  char* ws = (char*)d_ws;
  u16* xb  = (u16*)ws;                                  // [M,E] bf16
  u16* ctx = xb;                                        // reused after QKV
  u16* wqb = (u16*)(ws + (size_t)M * E * 2);
  u16* wkb = wqb + (size_t)E * E;
  u16* wvb = wkb + (size_t)E * E;
  u16* wob = wvb + (size_t)E * E;
  u16* Qb  = wob + (size_t)E * E;
  u16* Kb  = Qb + (size_t)M * E;
  u16* Vb  = Kb + (size_t)M * E;

  cvt_f32_bf16<<<dim3((M * E / 4 + 255) / 256), 256, 0, stream>>>(x, xb, M * E / 4);
  cvt_f32_bf16<<<dim3((E * E / 4 + 255) / 256), 256, 0, stream>>>(Wq, wqb, E * E / 4);
  cvt_f32_bf16<<<dim3((E * E / 4 + 255) / 256), 256, 0, stream>>>(Wk, wkb, E * E / 4);
  cvt_f32_bf16<<<dim3((E * E / 4 + 255) / 256), 256, 0, stream>>>(Wv, wvb, E * E / 4);
  cvt_f32_bf16<<<dim3((E * E / 4 + 255) / 256), 256, 0, stream>>>(Wo, wob, E * E / 4);

  dim3 gg(E / 64, M / 64);
  gemm_bf16w<false><<<gg, 128, 0, stream>>>(xb, wqb, Qb, nullptr, nullptr, M, E, E);
  gemm_bf16w<false><<<gg, 128, 0, stream>>>(xb, wkb, Kb, nullptr, nullptr, M, E, E);
  gemm_bf16w<false><<<gg, 128, 0, stream>>>(xb, wvb, Vb, nullptr, nullptr, M, E, E);

  mha_attn<<<dim3(SEQ / 128, HEADS, BATCH), 256, 0, stream>>>(Qb, Kb, Vb, ctx);

  gemm_bf16w<true><<<gg, 128, 0, stream>>>(ctx, wob, nullptr, (float*)d_out, bo, M, E, E);
}
